// PointNetPlusPlusSegmentation_60318520705091
// MI455X (gfx1250) — matrix operand order repaired and checked by reading, not compiled, not run on hardware
//
#include <hip/hip_runtime.h>
#include <hip/hip_bf16.h>

// ---------------------------------------------------------------------------
// PointNet++ segmentation forward pass for gfx1250 (MI455X).
// MLP layers = bf16 WMMA GEMMs (v_wmma_f32_16x16x32_bf16, f32 accumulate).
// Activations stored bf16 with K padded to 32 so the GEMM inner loop is
// branch-free; each wave register-blocks 4 N-tiles (1 A-frag -> 4 WMMAs).
// All B fragments are preloaded per K-step so loads clause together and
// WMMAs overlap outstanding loads via staged s_wait_loadcnt.
// ---------------------------------------------------------------------------

typedef __bf16 bf16;
typedef __attribute__((ext_vector_type(16))) __bf16 v16bf;
typedef __attribute__((ext_vector_type(8)))  __bf16 bf16x8;
typedef __attribute__((ext_vector_type(8)))  float  v8f;

#define EPS_BN 1e-5f
#define NT 4                       // N-tiles (16 cols each) per wave

static inline int ceil32(int x) { return (x + 31) & ~31; }
static inline int ceil64(int x) { return (x + 63) & ~63; }

// ---------------------------------------------------------------------------
// WMMA GEMM: Y[M, Cout](f32) = X[M, Kpad](bf16) * Wb[CoutP, Kpad](bf16)^T + b
// Requirements: M % 128 == 0, Kpad % 32 == 0, CoutP % 64 == 0, zero pads.
// Each wave: one 16-row M tile x NT 16-col N tiles. 8 waves / block.
// A frag: lane (mr,h): K chunk [k0+16h, +16).
// B frag: lane (nr,h): K chunk [k0+16h, +16).
// D: VGPR r -> row (mtile + r + 8h), col (ntile + mr).
// ---------------------------------------------------------------------------
__global__ void wmma_gemm_bias(const bf16* __restrict__ X, int Kpad,
                               const bf16* __restrict__ Wb,
                               const float* __restrict__ bias,
                               float* __restrict__ Y,
                               int M, int Cout) {
    const int lane   = threadIdx.x & 31;
    const int wave   = threadIdx.x >> 5;
    const int ntile0 = blockIdx.x * (16 * NT);
    const int mtile  = (blockIdx.y * 8 + wave) * 16;
    const int mr = lane & 15;
    const int h  = lane >> 4;

    const bf16* Xr = X  + (size_t)(mtile + mr) * Kpad + 16 * h;
    const bf16* Wr = Wb + (size_t)(ntile0 + mr) * Kpad + 16 * h;
    const size_t wstep = (size_t)16 * Kpad;   // 16 weight rows per N tile

    v8f acc[NT] = {};
#pragma unroll 2
    for (int k0 = 0; k0 < Kpad; k0 += 32) {
        // ---- issue all loads for this K-step up front (clause-friendly) ----
        const bf16x8 c0 = *(const bf16x8*)(Xr + k0);
        const bf16x8 c1 = *(const bf16x8*)(Xr + k0 + 8);
        v16bf bb[NT];
#pragma unroll
        for (int t = 0; t < NT; ++t)
            bb[t] = *(const v16bf*)(Wr + (size_t)t * wstep + k0);

        const v16bf a = __builtin_shufflevector(
            c0, c1, 0, 1, 2, 3, 4, 5, 6, 7, 8, 9, 10, 11, 12, 13, 14, 15);
#pragma unroll
        for (int t = 0; t < NT; ++t)
            acc[t] = __builtin_amdgcn_wmma_f32_16x16x32_bf16(
                false, a, false, bb[t], (short)0, acc[t], false, false);
    }

#pragma unroll
    for (int t = 0; t < NT; ++t) {
        const int col = ntile0 + t * 16 + mr;
        if (col < Cout) {
            const float bv = bias[col];
#pragma unroll
            for (int r = 0; r < 8; ++r) {
                const int row = mtile + r + 8 * h;
                Y[(long)row * Cout + col] = acc[t][r] + bv;
            }
        }
    }
}

// ---------------------------------------------------------------------------
// Pack f32 weights [Cout, K] -> bf16 [CoutP, Kpad], zero padded.
// ---------------------------------------------------------------------------
__global__ void pack_weights_kernel(const float* __restrict__ W, int Cout, int K,
                                    bf16* __restrict__ out, int Kpad, int CoutP) {
    const long total = (long)CoutP * Kpad;
    for (long i = (long)blockIdx.x * blockDim.x + threadIdx.x; i < total;
         i += (long)gridDim.x * blockDim.x) {
        const int k = (int)(i % Kpad);
        const int c = (int)(i / Kpad);
        const float v = (c < Cout && k < K) ? W[(long)c * K + k] : 0.f;
        out[i] = (bf16)v;
    }
}

// ---------------------------------------------------------------------------
// Per-channel mean / variance over M rows (one block per channel).
// Deterministic tree reduction (no float atomics).
// ---------------------------------------------------------------------------
__global__ void stats_kernel(const float* __restrict__ Y, long M, int C,
                             float* __restrict__ mean, float* __restrict__ var) {
    __shared__ float ssum[256];
    __shared__ float ssq[256];
    const int c = blockIdx.x;
    float s = 0.f, q = 0.f;
    for (long r = threadIdx.x; r < M; r += blockDim.x) {
        const float v = Y[r * C + c];
        s += v; q += v * v;
    }
    ssum[threadIdx.x] = s; ssq[threadIdx.x] = q;
    __syncthreads();
    for (int st = 128; st > 0; st >>= 1) {
        if (threadIdx.x < st) {
            ssum[threadIdx.x] += ssum[threadIdx.x + st];
            ssq[threadIdx.x]  += ssq[threadIdx.x + st];
        }
        __syncthreads();
    }
    if (threadIdx.x == 0) {
        const float m = ssum[0] / (float)M;
        mean[c] = m;
        var[c]  = fmaxf(ssq[0] / (float)M - m * m, 0.f);
    }
}

// BN + ReLU, f32 in (stride C) -> bf16 out (stride Cpad, pad zeroed)
__global__ void bn_relu_store_kernel(const float* __restrict__ Y, long M,
                                     int C, int Cpad,
                                     const float* __restrict__ mean,
                                     const float* __restrict__ var,
                                     const float* __restrict__ gamma,
                                     const float* __restrict__ beta,
                                     bf16* __restrict__ X) {
    const long total = M * Cpad;
    for (long i = (long)blockIdx.x * blockDim.x + threadIdx.x; i < total;
         i += (long)gridDim.x * blockDim.x) {
        const int c = (int)(i % Cpad);
        float v = 0.f;
        if (c < C) {
            const long r = i / Cpad;
            const float y = Y[r * C + c];
            v = fmaxf((y - mean[c]) * rsqrtf(var[c] + EPS_BN) * gamma[c] + beta[c],
                      0.f);
        }
        X[i] = (bf16)v;
    }
}

// ---------------------------------------------------------------------------
// Farthest point sampling (one block per batch).
// ---------------------------------------------------------------------------
__global__ void fps_kernel(const float* __restrict__ pts, int N, int stride,
                           int npoint, float* __restrict__ dist,
                           int* __restrict__ out_idx,
                           float* __restrict__ new_xyz) {
    __shared__ float sval[256];
    __shared__ int   sidx[256];
    __shared__ int   s_far;
    __shared__ float s_c[3];
    const int b = blockIdx.x;
    const int t = threadIdx.x;
    const float* P = pts + (long)b * N * stride;
    float* D = dist + (long)b * N;

    for (int j = t; j < N; j += blockDim.x) D[j] = 1e10f;
    if (t == 0) s_far = 0;
    __syncthreads();

    for (int i = 0; i < npoint; ++i) {
        if (t == 0) {
            const int far = s_far;
            out_idx[b * npoint + i] = far;
            const float cx = P[(long)far * stride + 0];
            const float cy = P[(long)far * stride + 1];
            const float cz = P[(long)far * stride + 2];
            s_c[0] = cx; s_c[1] = cy; s_c[2] = cz;
            new_xyz[((long)b * npoint + i) * 3 + 0] = cx;
            new_xyz[((long)b * npoint + i) * 3 + 1] = cy;
            new_xyz[((long)b * npoint + i) * 3 + 2] = cz;
        }
        __syncthreads();
        const float cx = s_c[0], cy = s_c[1], cz = s_c[2];
        float best = -1.f; int bidx = 0;
        for (int j = t; j < N; j += blockDim.x) {
            const float dx = P[(long)j * stride + 0] - cx;
            const float dy = P[(long)j * stride + 1] - cy;
            const float dz = P[(long)j * stride + 2] - cz;
            const float d = dx * dx + dy * dy + dz * dz;
            const float nd = fminf(D[j], d);
            D[j] = nd;
            if (nd > best) { best = nd; bidx = j; }
        }
        sval[t] = best; sidx[t] = bidx;
        __syncthreads();
        for (int st = 128; st > 0; st >>= 1) {
            if (t < st) {
                if (sval[t + st] > sval[t] ||
                    (sval[t + st] == sval[t] && sidx[t + st] < sidx[t])) {
                    sval[t] = sval[t + st]; sidx[t] = sidx[t + st];
                }
            }
            __syncthreads();
        }
        if (t == 0) s_far = sidx[0];
        __syncthreads();
    }
}

// ---------------------------------------------------------------------------
// Ball query: first nsample indices (ascending) with d <= r^2, padded.
// ---------------------------------------------------------------------------
__global__ void ball_query_kernel(const float* __restrict__ xyz, int N, int stride,
                                  const float* __restrict__ new_xyz, int S,
                                  float r2, int nsample,
                                  int* __restrict__ out_idx, int B) {
    const int gid = blockIdx.x * blockDim.x + threadIdx.x;
    if (gid >= B * S) return;
    const int b = gid / S;
    const float* P = xyz + (long)b * N * stride;
    const float qx = new_xyz[(long)gid * 3 + 0];
    const float qy = new_xyz[(long)gid * 3 + 1];
    const float qz = new_xyz[(long)gid * 3 + 2];
    int* out = out_idx + (long)gid * nsample;
    int cnt = 0;
    for (int j = 0; j < N && cnt < nsample; ++j) {
        const float dx = P[(long)j * stride + 0] - qx;
        const float dy = P[(long)j * stride + 1] - qy;
        const float dz = P[(long)j * stride + 2] - qz;
        if (dx * dx + dy * dy + dz * dz <= r2) out[cnt++] = j;
    }
    const int first = (cnt > 0) ? out[0] : 0;
    for (int k = cnt; k < nsample; ++k) out[k] = first;
}

// ---------------------------------------------------------------------------
// Grouped tensor [B,S,K, Cpad] (bf16): (xyz[idx]-new_xyz) ++ points[idx].
// ---------------------------------------------------------------------------
__global__ void group_kernel(const float* __restrict__ xyz, int N, int xstride,
                             const float* __restrict__ pts, int pdim,
                             int pstride, int poffset,
                             const float* __restrict__ new_xyz,
                             const int* __restrict__ idx,
                             int S, int K, bf16* __restrict__ out, int B,
                             int Cpad) {
    const int C = 3 + pdim;
    const long total = (long)B * S * K * Cpad;
    for (long i = (long)blockIdx.x * blockDim.x + threadIdx.x; i < total;
         i += (long)gridDim.x * blockDim.x) {
        const int c = (int)(i % Cpad);
        long r = i / Cpad;
        const int k = (int)(r % K);  r /= K;
        const int s = (int)(r % S);
        const int b = (int)(r / S);
        float v = 0.f;
        if (c < C) {
            const int j = idx[((long)b * S + s) * K + k];
            if (c < 3)
                v = xyz[((long)b * N + j) * xstride + c] -
                    new_xyz[((long)b * S + s) * 3 + c];
            else
                v = pts[((long)b * N + j) * pstride + poffset + (c - 3)];
        }
        out[i] = (bf16)v;
    }
}

// Max-pool over the group dim: out[bs,c] (f32) = max_k in[(bs*K+k)*C+c] (bf16)
__global__ void maxpool_kernel(const bf16* __restrict__ in, int K, int C,
                               float* __restrict__ out, long rows) {
    const long total = rows * C;
    for (long i = (long)blockIdx.x * blockDim.x + threadIdx.x; i < total;
         i += (long)gridDim.x * blockDim.x) {
        const int c = (int)(i % C);
        const long bs = i / C;
        float m = -3.4e38f;
        for (int k = 0; k < K; ++k)
            m = fmaxf(m, (float)in[((long)bs * K + k) * C + c]);
        out[i] = m;
    }
}

// group_all: bf16 [B,S,Cpad] = l2_xyz ++ l2_p
__global__ void group_all_kernel(const float* __restrict__ xyz,
                                 const float* __restrict__ pts, int pdim,
                                 int S, bf16* __restrict__ out, int B, int Cpad) {
    const int C = 3 + pdim;
    const long total = (long)B * S * Cpad;
    for (long i = (long)blockIdx.x * blockDim.x + threadIdx.x; i < total;
         i += (long)gridDim.x * blockDim.x) {
        const int c = (int)(i % Cpad);
        const long bs = i / Cpad;
        float v = 0.f;
        if (c < 3) v = xyz[bs * 3 + c];
        else if (c < C) v = pts[bs * pdim + (c - 3)];
        out[i] = (bf16)v;
    }
}

// concat points1 ++ broadcast(points2[b]) (M==1 feature propagation), bf16 out
__global__ void concat_bcast_kernel(const float* __restrict__ p1, int c1,
                                    const float* __restrict__ p2, int c2,
                                    int S, bf16* __restrict__ out, int B,
                                    int Cpad) {
    const int C = c1 + c2;
    const long total = (long)B * S * Cpad;
    for (long i = (long)blockIdx.x * blockDim.x + threadIdx.x; i < total;
         i += (long)gridDim.x * blockDim.x) {
        const int c = (int)(i % Cpad);
        const long bs = i / Cpad;
        const int b = (int)(bs / S);
        float v = 0.f;
        if (c < c1) v = p1[bs * c1 + c];
        else if (c < C) v = p2[(long)b * c2 + (c - c1)];
        out[i] = (bf16)v;
    }
}

// 3-nearest-neighbors: idx + normalized inverse-distance weights
__global__ void knn3_kernel(const float* __restrict__ xyz1, int N1, int s1,
                            const float* __restrict__ xyz2, int N2,
                            int* __restrict__ idx3, float* __restrict__ w3,
                            int B) {
    const int gid = blockIdx.x * blockDim.x + threadIdx.x;
    if (gid >= B * N1) return;
    const int b = gid / N1;
    const float px = xyz1[(long)gid * s1 + 0];
    const float py = xyz1[(long)gid * s1 + 1];
    const float pz = xyz1[(long)gid * s1 + 2];
    const float* Q = xyz2 + (long)b * N2 * 3;
    float d0 = 3.4e38f, d1 = 3.4e38f, d2 = 3.4e38f;
    int i0 = 0, i1 = 0, i2 = 0;
    for (int j = 0; j < N2; ++j) {
        const float dx = Q[j * 3 + 0] - px;
        const float dy = Q[j * 3 + 1] - py;
        const float dz = Q[j * 3 + 2] - pz;
        const float d = dx * dx + dy * dy + dz * dz;
        if (d < d0)      { d2 = d1; i2 = i1; d1 = d0; i1 = i0; d0 = d; i0 = j; }
        else if (d < d1) { d2 = d1; i2 = i1; d1 = d;  i1 = j; }
        else if (d < d2) { d2 = d;  i2 = j; }
    }
    const float r0 = 1.f / (d0 + 1e-8f);
    const float r1 = 1.f / (d1 + 1e-8f);
    const float r2 = 1.f / (d2 + 1e-8f);
    const float s = r0 + r1 + r2;
    idx3[(long)gid * 3 + 0] = i0;
    idx3[(long)gid * 3 + 1] = i1;
    idx3[(long)gid * 3 + 2] = i2;
    w3[(long)gid * 3 + 0] = r0 / s;
    w3[(long)gid * 3 + 1] = r1 / s;
    w3[(long)gid * 3 + 2] = r2 / s;
}

// out[bn,:] (bf16, Cpad) = points1[bn] ++ sum_t w[t]*points2[b, idx[t]]
__global__ void interp_concat_kernel(const float* __restrict__ p1, int c1,
                                     int p1stride, int p1off,
                                     const float* __restrict__ p2, int c2, int N2,
                                     const int* __restrict__ idx3,
                                     const float* __restrict__ w3,
                                     int N1, bf16* __restrict__ out, int B,
                                     int Cpad) {
    const int C = c1 + c2;
    const long total = (long)B * N1 * Cpad;
    for (long i = (long)blockIdx.x * blockDim.x + threadIdx.x; i < total;
         i += (long)gridDim.x * blockDim.x) {
        const int c = (int)(i % Cpad);
        const long bn = i / Cpad;
        float v = 0.f;
        if (c < c1) {
            v = p1[bn * p1stride + p1off + c];
        } else if (c < C) {
            const int cc = c - c1;
            const int b = (int)(bn / N1);
            const int* id = idx3 + bn * 3;
            const float* w = w3 + bn * 3;
#pragma unroll
            for (int t = 0; t < 3; ++t)
                v += w[t] * p2[((long)b * N2 + id[t]) * c2 + cc];
        }
        out[i] = (bf16)v;
    }
}

// max over S points per batch: out[b,c] (f32) = max_p in[(b*S+p)*C+c] (bf16)
__global__ void max_points_kernel(const bf16* __restrict__ in, int S, int C,
                                  float* __restrict__ out, int B) {
    const long total = (long)B * C;
    for (long i = (long)blockIdx.x * blockDim.x + threadIdx.x; i < total;
         i += (long)gridDim.x * blockDim.x) {
        const int c = (int)(i % C);
        const int b = (int)(i / C);
        float m = -3.4e38f;
        for (int p = 0; p < S; ++p)
            m = fmaxf(m, (float)in[((long)b * S + p) * C + c]);
        out[i] = m;
    }
}

// bf16 (contiguous) -> f32 copy
__global__ void bf16_to_f32_kernel(const bf16* __restrict__ in,
                                   float* __restrict__ out, long total) {
    for (long i = (long)blockIdx.x * blockDim.x + threadIdx.x; i < total;
         i += (long)gridDim.x * blockDim.x)
        out[i] = (float)in[i];
}

// ---------------------------------------------------------------------------
// Host-side orchestration
// ---------------------------------------------------------------------------
namespace {

struct Layer {
    const bf16*  Wb;     // packed bf16 [CoutP(x64), Kpad(x32)]
    const float* bias;
    const float* gamma;
    const float* beta;
    int cout;
};

inline int nblk(long total) {
    long nb = (total + 255) / 256;
    return (int)(nb < 4096 ? (nb > 0 ? nb : 1) : 4096);
}

inline void run_mlp(hipStream_t st, const Layer* Ls, int nL,
                    bf16* X, bf16* Xalt, float* Yf,
                    long M, int Kin, float* mean, float* var,
                    bf16** outBuf, int* outC) {
    int K = Kin;
    for (int l = 0; l < nL; ++l) {
        const int C  = Ls[l].cout;
        const int Kp = ceil32(K);
        dim3 grid(ceil64(C) / 64, (unsigned)(M / 128));
        wmma_gemm_bias<<<grid, 256, 0, st>>>(X, Kp, Ls[l].Wb, Ls[l].bias, Yf,
                                             (int)M, C);
        stats_kernel<<<C, 256, 0, st>>>(Yf, M, C, mean, var);
        const int Cp = ceil32(C);
        bn_relu_store_kernel<<<nblk(M * Cp), 256, 0, st>>>(
            Yf, M, C, Cp, mean, var, Ls[l].gamma, Ls[l].beta, Xalt);
        bf16* t = X; X = Xalt; Xalt = t;
        K = C;
    }
    *outBuf = X;
    *outC = K;
}

} // namespace

extern "C" void kernel_launch(void* const* d_in, const int* in_sizes, int n_in,
                              void* d_out, int out_size, void* d_ws, size_t ws_size,
                              hipStream_t stream) {
    (void)in_sizes; (void)n_in; (void)out_size; (void)ws_size;

    constexpr int B = 16, N = 4096;
    const float* x = (const float*)d_in[0];

    // ---- workspace cursor ----
    char* ws = (char*)d_ws;
    size_t off = 0;
    auto alloc = [&](size_t bytes) -> void* {
        void* q = ws + off;
        off = (off + bytes + 255) & ~(size_t)255;
        return q;
    };

    // ---- parameters: setup_inputs() insertion order, packed to bf16 ----
    int p = 1;
    auto prepLayer = [&](int cout, int kin) {
        Layer L;
        const float* W = (const float*)d_in[p++];
        L.bias  = (const float*)d_in[p++];
        L.gamma = (const float*)d_in[p++];
        L.beta  = (const float*)d_in[p++];
        L.cout  = cout;
        const int Kp = ceil32(kin), Cp = ceil64(cout);
        bf16* wb = (bf16*)alloc((size_t)Cp * Kp * sizeof(bf16));
        pack_weights_kernel<<<nblk((long)Cp * Kp), 256, 0, stream>>>(
            W, cout, kin, wb, Kp, Cp);
        L.Wb = wb;
        return L;
    };
    Layer sa1[3]   = { prepLayer(64, 9),    prepLayer(64, 64),   prepLayer(128, 64)   };
    Layer sa2[3]   = { prepLayer(128, 131), prepLayer(128, 128), prepLayer(256, 128)  };
    Layer sa3[3]   = { prepLayer(256, 259), prepLayer(512, 256), prepLayer(1024, 512) };
    Layer fp3[2]   = { prepLayer(256, 1280), prepLayer(256, 256) };
    Layer fp2[2]   = { prepLayer(256, 384),  prepLayer(128, 256) };
    Layer fp1[3]   = { prepLayer(128, 134), prepLayer(128, 128), prepLayer(128, 128) };
    Layer head1[1] = { prepLayer(128, 128) };
    Layer conv2;
    {
        const float* W = (const float*)d_in[p++];
        conv2.bias = (const float*)d_in[p++];
        conv2.gamma = nullptr; conv2.beta = nullptr;
        conv2.cout = 13;
        bf16* wb = (bf16*)alloc((size_t)64 * 128 * sizeof(bf16));
        pack_weights_kernel<<<nblk(64L * 128), 256, 0, stream>>>(W, 13, 128, wb,
                                                                 128, 64);
        conv2.Wb = wb;
    }

    // ---- activation / intermediate buffers ----
    bf16*  bufA    = (bf16*)alloc((size_t)20971520 * sizeof(bf16)); // >=131072*160
    bf16*  bufB    = (bf16*)alloc((size_t)33554432 * sizeof(bf16)); // >=262144*128
    float* bufY    = (float*)alloc((size_t)33554432 * sizeof(float));
    float* dist    = (float*)alloc((size_t)B * N * 4);
    float* l1_xyz  = (float*)alloc((size_t)B * 512 * 3 * 4);
    float* l2_xyz  = (float*)alloc((size_t)B * 128 * 3 * 4);
    float* l1_p    = (float*)alloc((size_t)B * 512 * 128 * 4);
    float* l2_p    = (float*)alloc((size_t)B * 128 * 256 * 4);
    float* l3_p    = (float*)alloc((size_t)B * 1024 * 4);
    float* knn_w   = (float*)alloc((size_t)B * N * 3 * 4);
    float* meanb   = (float*)alloc((size_t)1024 * 4);
    float* varb    = (float*)alloc((size_t)1024 * 4);
    int*   fps_idx = (int*)alloc((size_t)B * 512 * 4);
    int*   ballidx = (int*)alloc((size_t)B * 512 * 32 * 4);
    int*   knn_idx = (int*)alloc((size_t)B * N * 3 * 4);

    bf16* outBuf; int outC;

    // ================= SA1: 4096 -> 512 pts, r=0.2, K=32 =================
    fps_kernel<<<B, 256, 0, stream>>>(x, N, 9, 512, dist, fps_idx, l1_xyz);
    ball_query_kernel<<<(B * 512 + 255) / 256, 256, 0, stream>>>(
        x, N, 9, l1_xyz, 512, 0.04f, 32, ballidx, B);
    {
        const long M1 = (long)B * 512 * 32;               // 262144
        group_kernel<<<nblk(M1 * 32), 256, 0, stream>>>(
            x, N, 9, x, 6, 9, 3, l1_xyz, ballidx, 512, 32, bufA, B, 32);
        run_mlp(stream, sa1, 3, bufA, bufB, bufY, M1, 9, meanb, varb,
                &outBuf, &outC);
        maxpool_kernel<<<nblk((long)B * 512 * 128), 256, 0, stream>>>(
            outBuf, 32, 128, l1_p, (long)B * 512);
    }

    // ================= SA2: 512 -> 128 pts, r=0.4, K=64 =================
    fps_kernel<<<B, 256, 0, stream>>>(l1_xyz, 512, 3, 128, dist, fps_idx, l2_xyz);
    ball_query_kernel<<<(B * 128 + 255) / 256, 256, 0, stream>>>(
        l1_xyz, 512, 3, l2_xyz, 128, 0.16f, 64, ballidx, B);
    {
        const long M2 = (long)B * 128 * 64;               // 131072
        group_kernel<<<nblk(M2 * 160), 256, 0, stream>>>(
            l1_xyz, 512, 3, l1_p, 128, 128, 0, l2_xyz, ballidx, 128, 64,
            bufA, B, 160);
        run_mlp(stream, sa2, 3, bufA, bufB, bufY, M2, 131, meanb, varb,
                &outBuf, &outC);
        maxpool_kernel<<<nblk((long)B * 128 * 256), 256, 0, stream>>>(
            outBuf, 64, 256, l2_p, (long)B * 128);
    }

    // ================= SA3 (group all): 128 pts -> 1 =================
    {
        const long M3 = (long)B * 128;                    // 2048
        group_all_kernel<<<nblk(M3 * 288), 256, 0, stream>>>(
            l2_xyz, l2_p, 256, 128, bufA, B, 288);
        run_mlp(stream, sa3, 3, bufA, bufB, bufY, M3, 259, meanb, varb,
                &outBuf, &outC);
        max_points_kernel<<<nblk((long)B * 1024), 256, 0, stream>>>(
            outBuf, 128, 1024, l3_p, B);
    }

    // ================= FP3: broadcast l3_p -> 128 pts =================
    {
        const long M3 = (long)B * 128;
        concat_bcast_kernel<<<nblk(M3 * 1280), 256, 0, stream>>>(
            l2_p, 256, l3_p, 1024, 128, bufA, B, 1280);
        run_mlp(stream, fp3, 2, bufA, bufB, bufY, M3, 1280, meanb, varb,
                &outBuf, &outC);
        bf16_to_f32_kernel<<<nblk(M3 * 256), 256, 0, stream>>>(outBuf, l2_p,
                                                               M3 * 256);
    }

    // ================= FP2: 128 -> 512 pts (3-NN) =================
    {
        const long M2 = (long)B * 512;
        knn3_kernel<<<(B * 512 + 255) / 256, 256, 0, stream>>>(
            l1_xyz, 512, 3, l2_xyz, 128, knn_idx, knn_w, B);
        interp_concat_kernel<<<nblk(M2 * 384), 256, 0, stream>>>(
            l1_p, 128, 128, 0, l2_p, 256, 128, knn_idx, knn_w, 512, bufA, B, 384);
        run_mlp(stream, fp2, 2, bufA, bufB, bufY, M2, 384, meanb, varb,
                &outBuf, &outC);
        bf16_to_f32_kernel<<<nblk(M2 * 128), 256, 0, stream>>>(outBuf, l1_p,
                                                               M2 * 128);
    }

    // ================= FP1: 512 -> 4096 pts (3-NN) =================
    const long M0 = (long)B * N;                          // 65536
    {
        knn3_kernel<<<(B * N + 255) / 256, 256, 0, stream>>>(
            x, N, 9, l1_xyz, 512, knn_idx, knn_w, B);
        interp_concat_kernel<<<nblk(M0 * 160), 256, 0, stream>>>(
            x, 6, 9, 3, l1_p, 128, 512, knn_idx, knn_w, N, bufA, B, 160);
        run_mlp(stream, fp1, 3, bufA, bufB, bufY, M0, 134, meanb, varb,
                &outBuf, &outC);
    }

    // ================= head + logits =================
    {
        run_mlp(stream, head1, 1, outBuf, (outBuf == bufA) ? bufB : bufA, bufY,
                M0, 128, meanb, varb, &outBuf, &outC);
        dim3 grid(1, (unsigned)(M0 / 128));
        wmma_gemm_bias<<<grid, 256, 0, stream>>>(outBuf, 128, conv2.Wb,
                                                 conv2.bias, (float*)d_out,
                                                 (int)M0, 13);
    }
}
